// MLP_Diag_20753281974772
// MI455X (gfx1250) — matrix operand order repaired and checked by reading, not compiled, not run on hardware
//
#include <hip/hip_runtime.h>
#include <hip/hip_bf16.h>
#include <math.h>

typedef _Float16 v8h  __attribute__((ext_vector_type(8)));
typedef _Float16 v16h __attribute__((ext_vector_type(16)));
typedef float    v8f  __attribute__((ext_vector_type(8)));

#define NN   8192
#define DD   256
#define KEEP 31      // k+1 = 30+1 entries kept per row

// ---------------------------------------------------------------------------
// Kernel 1: h = relu(x*w0)*w1 ; emb = h / max(||h||, 1e-12) ; store fp16
// One block (256 threads == D) per row.
// ---------------------------------------------------------------------------
__global__ __launch_bounds__(DD) void embed_kernel(const float* __restrict__ feat,
                                                   const float* __restrict__ w,
                                                   _Float16* __restrict__ E) {
    const int row = blockIdx.x;
    const int tid = threadIdx.x;

    float h = feat[(size_t)row * DD + tid] * w[tid];
    h = fmaxf(h, 0.0f);
    h *= w[DD + tid];

    __shared__ float red[DD];
    red[tid] = h * h;
    __syncthreads();
#pragma unroll
    for (int s = DD / 2; s > 0; s >>= 1) {
        if (tid < s) red[tid] += red[tid + s];
        __syncthreads();
    }
    const float norm = fmaxf(sqrtf(red[0]), 1e-12f);
    E[(size_t)row * DD + tid] = (_Float16)(h / norm);
}

// ---------------------------------------------------------------------------
// Kernel 2: sim = E @ E^T via v_wmma_f32_16x16x32_f16.
// Block = 128 threads (4 waves), each wave computes a 64x64 tile (4x4 WMMA
// accumulators). E (4 MB fp16) is L2-resident, fragments loaded directly from
// global per the CDNA5 VGPR layouts (symmetry => A and B come from the same
// row-major matrix, no transpose loads needed).
// ---------------------------------------------------------------------------
__global__ __launch_bounds__(128) void simgemm_kernel(const _Float16* __restrict__ E,
                                                      float* __restrict__ out) {
    const int tid  = threadIdx.x;
    const int lane = tid & 31;
    const int wv   = tid >> 5;         // wave 0..3 -> 2x2 arrangement
    const int lm   = lane & 15;
    const int half = lane >> 4;

    const int rowBase = blockIdx.y * 128 + (wv >> 1) * 64;
    const int colBase = blockIdx.x * 128 + (wv & 1) * 64;

    v8f acc[4][4];
#pragma unroll
    for (int i = 0; i < 4; ++i)
#pragma unroll
        for (int j = 0; j < 4; ++j)
            acc[i][j] = (v8f){0.f, 0.f, 0.f, 0.f, 0.f, 0.f, 0.f, 0.f};

    for (int kb = 0; kb < DD; kb += 32) {
        v16h A[4], B[4];
        // A fragment: lane holds row (rowBase+16i+lm),
        // K = kb + {8h..8h+7} and kb + {16+8h..23+8h}  (two 16B chunks)
#pragma unroll
        for (int i = 0; i < 4; ++i) {
            const _Float16* p = E + (size_t)(rowBase + 16 * i + lm) * DD + kb + 8 * half;
            v8h c0 = *(const v8h*)(p);
            v8h c1 = *(const v8h*)(p + 16);
#pragma unroll
            for (int e = 0; e < 8; ++e) { A[i][e] = c0[e]; A[i][e + 8] = c1[e]; }
        }
        // B fragment (symmetric operand): lane holds "column" row (colBase+16j+lm),
        // K = kb + {8h..8h+7} and kb + {16+8h..23+8h}  (two 16B chunks)
#pragma unroll
        for (int j = 0; j < 4; ++j) {
            const _Float16* p = E + (size_t)(colBase + 16 * j + lm) * DD + kb + 8 * half;
            v8h c0 = *(const v8h*)(p);
            v8h c1 = *(const v8h*)(p + 16);
#pragma unroll
            for (int e = 0; e < 8; ++e) { B[j][e] = c0[e]; B[j][e + 8] = c1[e]; }
        }
#pragma unroll
        for (int i = 0; i < 4; ++i)
#pragma unroll
            for (int j = 0; j < 4; ++j)
                acc[i][j] = __builtin_amdgcn_wmma_f32_16x16x32_f16(
                    /*neg_a=*/false, A[i], /*neg_b=*/false, B[j],
                    /*c_mod=*/(short)0, acc[i][j],
                    /*reuse_a=*/false, /*reuse_b=*/false);
    }

    // C layout: lane holds column n = colBase+16j+lm, VGPR t -> row 8*half + t
#pragma unroll
    for (int i = 0; i < 4; ++i)
#pragma unroll
        for (int j = 0; j < 4; ++j) {
            const int n  = colBase + 16 * j + lm;
            const int mB = rowBase + 16 * i + 8 * half;
#pragma unroll
            for (int t = 0; t < 8; ++t)
                out[(size_t)(mB + t) * NN + n] = acc[i][j][t];
        }
}

// ---------------------------------------------------------------------------
// Kernel 3: per-row keep top-(KEEP) entries, zero rest, relu. One block per
// row; 32 KB row cached in LDS; binary search the KEEP-th largest value.
// ---------------------------------------------------------------------------
__global__ __launch_bounds__(256) void topk_relu_kernel(float* __restrict__ sim) {
    const int row = blockIdx.x;
    const int tid = threadIdx.x;
    float* rp = sim + (size_t)row * NN;

    __shared__ float rowv[NN];       // 32 KB (LDS is 320 KB/WGP)
    __shared__ int   cnt[256];
    __shared__ float red[256];
    __shared__ float s_lo, s_hi;

    // load my contiguous 32 elements, tracking local max
    float lmax = -2.0f;
#pragma unroll
    for (int q = 0; q < 32; q += 4) {
        const int base = tid * 32 + q;
        float4 v = *(const float4*)(rp + base);
        *(float4*)(&rowv[base]) = v;
        lmax = fmaxf(lmax, fmaxf(fmaxf(v.x, v.y), fmaxf(v.z, v.w)));
    }
    red[tid] = lmax;
    __syncthreads();
#pragma unroll
    for (int s = 128; s > 0; s >>= 1) {
        if (tid < s) red[tid] = fmaxf(red[tid], red[tid + s]);
        __syncthreads();
    }
    if (tid == 0) { s_lo = -2.0f; s_hi = red[0] + 1.0f; }
    __syncthreads();

    // binary search threshold: invariant count(>=lo) >= KEEP, count(>=hi) < KEEP
    for (int it = 0; it < 32; ++it) {
        const float mid = 0.5f * (s_lo + s_hi);
        int c = 0;
#pragma unroll
        for (int q = 0; q < 32; ++q)
            c += (rowv[tid * 32 + q] >= mid) ? 1 : 0;
        cnt[tid] = c;
        __syncthreads();
#pragma unroll
        for (int s = 128; s > 0; s >>= 1) {
            if (tid < s) cnt[tid] += cnt[tid + s];
            __syncthreads();
        }
        if (tid == 0) {
            if (cnt[0] >= KEEP) s_lo = mid; else s_hi = mid;
        }
        __syncthreads();
    }
    const float thr = s_lo;

    // mask + relu, write back in place
#pragma unroll
    for (int q = 0; q < 32; q += 4) {
        const int base = tid * 32 + q;
        float4 v = *(const float4*)(&rowv[base]);
        v.x = (v.x >= thr) ? fmaxf(v.x, 0.0f) : 0.0f;
        v.y = (v.y >= thr) ? fmaxf(v.y, 0.0f) : 0.0f;
        v.z = (v.z >= thr) ? fmaxf(v.z, 0.0f) : 0.0f;
        v.w = (v.w >= thr) ? fmaxf(v.w, 0.0f) : 0.0f;
        *(float4*)(rp + base) = v;
    }
}

// ---------------------------------------------------------------------------
extern "C" void kernel_launch(void* const* d_in, const int* in_sizes, int n_in,
                              void* d_out, int out_size, void* d_ws, size_t ws_size,
                              hipStream_t stream) {
    const float* feat = (const float*)d_in[0];   // [8192, 256] f32
    const float* w    = (const float*)d_in[1];   // [2, 256]    f32
    float*       out  = (float*)d_out;           // [8192, 8192] f32
    _Float16*    E    = (_Float16*)d_ws;         // 4 MB fp16 embeddings

    embed_kernel<<<NN, DD, 0, stream>>>(feat, w, E);
    simgemm_kernel<<<dim3(NN / 128, NN / 128), 128, 0, stream>>>(E, out);
    topk_relu_kernel<<<NN, 256, 0, stream>>>(out);
}
